// WindowAttention_26164940767791
// MI455X (gfx1250) — compile-verified, matrix operand order checked by reading, not run on hardware
//
#include <hip/hip_runtime.h>

// ---------------------------------------------------------------------------
// Fused window attention for MI455X (gfx1250), wave32 + WMMA f16->f32.
//   x:(B,49,96) -> qkv GEMM -> per-head QK^T + bias -> softmax -> PV -> proj
// One workgroup (8 waves, 256 threads) processes WPB windows.
// Wave = (M-block 0..3) x (group 0..1): group splits the N-tile loops of the
// QKV / proj GEMMs and the head loop of the attention stage, giving 2 waves
// per SIMD32. All intermediates, f16 weight copies and the GEMM biases live
// in ~182KB of dynamic LDS; only x is read from and out written to HBM, with
// the next window's x prefetched (global_prefetch_b8) while computing.
// Wave-uniform ids go through readfirstlane so loop control is scalar and
// EXEC is structurally all-ones around every WMMA.
// ---------------------------------------------------------------------------

#define TOKENS   49
#define MP       64      // padded rows
#define CDIM     96
#define NQKV     288
#define NHEADS   3
#define HD       32
#define NWIN     16384
#define WPB      8
#define NTHREADS 256

typedef __attribute__((ext_vector_type(16))) _Float16 v16h;
typedef __attribute__((ext_vector_type(8)))  float    v8f;

union FragA { v16h v; _Float16 h[16]; };

// Generic 16x32 f16 operand loader in the CDNA5 A-operand layout:
//   lane L: m = row0 + (L&15), half = L>>4
//   vgpr r<4 : K = col0 + 8*half + 2r      (pair)
//   vgpr r>=4: K = col0 + 16 + 8*half + 2(r-4)
// For the B operand pass the row-major B^T source (rstride/cstride generic).
__device__ __forceinline__ v16h load_frag(const _Float16* base,
                                          int rstride, int cstride,
                                          int row0, int col0, int lane) {
  FragA f;
  const int m  = row0 + (lane & 15);
  const int hb = (lane >> 4) * 8;
#pragma unroll
  for (int r = 0; r < 8; ++r) {
    const int k = col0 + ((r < 4) ? (hb + 2 * r) : (16 + hb + 2 * (r - 4)));
    f.h[2 * r]     = base[m * rstride + k * cstride];
    f.h[2 * r + 1] = base[m * rstride + (k + 1) * cstride];
  }
  return f.v;
}

__device__ __forceinline__ v8f wmma16(v16h a, v16h b, v8f c) {
  return __builtin_amdgcn_wmma_f32_16x16x32_f16(false, a, false, b,
                                                (short)0, c, false, false);
}

// Dynamic-LDS partition (bytes)
#define OFF_WQKVT  0                      // f16 [288][96]   qkv_w^T   55296
#define OFF_WPROJT 55296                  // f16 [96][96]    proj_w^T  18432
#define OFF_BIAS   73728                  // f32 [3][49*49]  gathered   28812
#define OFF_XS     102544                 // f16 [64][96]    x tile     12288
#define OFF_QKVS   114832                 // f16 [64][288]   qkv        36864
#define OFF_PS     151696                 // f16 [2][64][64] probs      16384
#define OFF_AOS    168080                 // f16 [64][96]    attn out   12288
#define OFF_QKVB   180368                 // f32 [288]       qkv_b       1152
#define OFF_PROJB  181520                 // f32 [96]        proj_b       384
#define SMEM_BYTES 181904

extern "C" __global__ __launch_bounds__(NTHREADS)
void win_attn_fused(const float* __restrict__ x,
                    const float* __restrict__ qkv_w,
                    const float* __restrict__ qkv_b,
                    const float* __restrict__ proj_w,
                    const float* __restrict__ proj_b,
                    const float* __restrict__ bias_table,
                    const int*   __restrict__ rel_index,
                    float* __restrict__ out) {
  extern __shared__ char smem[];
  _Float16* wqkvT  = (_Float16*)(smem + OFF_WQKVT);
  _Float16* wprojT = (_Float16*)(smem + OFF_WPROJT);
  float*    biasF  = (float*)   (smem + OFF_BIAS);
  _Float16* xs     = (_Float16*)(smem + OFF_XS);
  _Float16* qkvs   = (_Float16*)(smem + OFF_QKVS);
  _Float16* aos    = (_Float16*)(smem + OFF_AOS);
  float*    qkvbL  = (float*)   (smem + OFF_QKVB);
  float*    projbL = (float*)   (smem + OFF_PROJB);

  const int tid  = threadIdx.x;
  const int lane = tid & 31;
  // wave-uniform ids as SGPRs -> scalar loop control, EXEC untouched
  const int wv   = __builtin_amdgcn_readfirstlane(tid >> 5);  // 0..7
  const int m0   = (wv & 3) * 16;   // M-block
  const int grp  = wv >> 2;         // work-split group 0/1
  const int half = lane >> 4;       // 0/1 half of wave
  const int nloc = lane & 15;
  const float scale = 0.17677669529663689f;   // 32^-0.5
  _Float16* ps = (_Float16*)(smem + OFF_PS) + grp * (MP * MP);

  // ---- one-time per-WG preload: weights (transposed f16), biases, rel-bias
  for (int i = tid; i < NQKV * CDIM; i += NTHREADS) {
    int n = i / CDIM, k = i - n * CDIM;
    wqkvT[i] = (_Float16)qkv_w[k * NQKV + n];
  }
  for (int i = tid; i < CDIM * CDIM; i += NTHREADS) {
    int n = i / CDIM, k = i - n * CDIM;
    wprojT[i] = (_Float16)proj_w[k * CDIM + n];
  }
  for (int i = tid; i < NHEADS * TOKENS * TOKENS; i += NTHREADS) {
    int h  = i / (TOKENS * TOKENS);
    int mn = i - h * (TOKENS * TOKENS);
    biasF[i] = bias_table[rel_index[mn] * NHEADS + h];
  }
  if (tid < NQKV) qkvbL[tid] = qkv_b[tid];
  if (tid < CDIM) projbL[tid] = proj_b[tid];
  __syncthreads();

  for (int wdx = 0; wdx < WPB; ++wdx) {
    const int b = blockIdx.x * WPB + wdx;

    // ---- prefetch next window's x into cache while we compute this one
    if (wdx + 1 < WPB) {
      const char* nx = (const char*)(x + ((size_t)(b + 1) * TOKENS) * CDIM);
      const int xbytes = TOKENS * CDIM * 4;
      for (int off = tid * 128; off < xbytes; off += NTHREADS * 128)
        __builtin_prefetch(nx + off, 0, 1);
    }

    // ---- stage x -> f16 LDS, zero-pad rows 49..63
    for (int i = tid; i < MP * CDIM; i += NTHREADS) {
      int m = i / CDIM, c = i - m * CDIM;
      float v = 0.f;
      if (m < TOKENS) v = x[((size_t)b * TOKENS + m) * CDIM + c];
      xs[i] = (_Float16)v;
    }
    __syncthreads();

    // ---- QKV GEMM: (64x96) x (96x288). Q columns get bias+scale folded in.
    // A fragments (this wave's 16 rows of x) are N-loop invariant: hoist.
    {
      v16h xa[CDIM / 32];
#pragma unroll
      for (int kt = 0; kt < CDIM / 32; ++kt)
        xa[kt] = load_frag(xs, CDIM, 1, m0, kt * 32, lane);
      for (int nt = grp; nt < NQKV / 16; nt += 2) {
        v8f acc = {};
#pragma unroll
        for (int kt = 0; kt < CDIM / 32; ++kt) {
          v16h bf = load_frag(wqkvT, CDIM, 1, nt * 16, kt * 32, lane);
          acc = wmma16(xa[kt], bf, acc);
        }
        const int   n  = nt * 16 + nloc;
        const float bb = qkvbL[n];
        const float sc = (nt < 6) ? scale : 1.0f;   // q slice is cols [0,96)
#pragma unroll
        for (int r = 0; r < 8; ++r)
          qkvs[(m0 + 8 * half + r) * NQKV + n] = (_Float16)((acc[r] + bb) * sc);
      }
    }
    __syncthreads();   // attention reads K,V rows written by all waves

    // ---- per-head attention; group 0: heads {0,2}, group 1: head {1}
    for (int h = grp; h < NHEADS; h += 2) {
      // S = Q K^T : one K-fragment (hd=32), 4 column tiles (padded 64)
      v16h aq = load_frag(qkvs, NQKV, 1, m0, h * HD, lane);
      v8f s[4];
#pragma unroll
      for (int t = 0; t < 4; ++t) {
        v16h bk = load_frag(qkvs, NQKV, 1, t * 16, CDIM + h * HD, lane);
        v8f z = {};
        s[t] = wmma16(aq, bk, z);
      }
      // bias + masked softmax per row (rows striped across 16-lane halves)
#pragma unroll
      for (int r = 0; r < 8; ++r) {
        const int m = m0 + 8 * half + r;
        float vals[4];
        float rowmax = -3.0e38f;
#pragma unroll
        for (int t = 0; t < 4; ++t) {
          const int n = t * 16 + nloc;
          float v = s[t][r];
          if (m < TOKENS && n < TOKENS)
            v += biasF[h * (TOKENS * TOKENS) + m * TOKENS + n];
          else
            v = -3.0e38f;
          vals[t] = v;
          rowmax = fmaxf(rowmax, v);
        }
        for (int off = 1; off < 16; off <<= 1)
          rowmax = fmaxf(rowmax, __shfl_xor(rowmax, off, 32));
        float sum = 0.f;
#pragma unroll
        for (int t = 0; t < 4; ++t) {
          float e = __expf(vals[t] - rowmax);   // masked -> exp(-huge) = 0
          vals[t] = e;
          sum += e;
        }
        for (int off = 1; off < 16; off <<= 1)
          sum += __shfl_xor(sum, off, 32);
        const float inv = 1.0f / sum;
#pragma unroll
        for (int t = 0; t < 4; ++t)
          ps[m * MP + t * 16 + nloc] = (_Float16)(vals[t] * inv);
      }
      // O = P V : K dim = 64 padded tokens, 2 d-tiles of 16.
      // A fragments of P are d-loop invariant: hoist (wave-local LDS RAW is
      // in-order, no barrier needed -- only this wave touches these rows).
      v16h ap[2];
#pragma unroll
      for (int kt = 0; kt < 2; ++kt)
        ap[kt] = load_frag(ps, MP, 1, m0, kt * 32, lane);
#pragma unroll
      for (int dt = 0; dt < 2; ++dt) {
        v8f acc = {};
#pragma unroll
        for (int kt = 0; kt < 2; ++kt) {
          // B^T[d][n] = V[n][d] : rows d contiguous, cols n stride NQKV
          v16h bv = load_frag(qkvs + (2 * CDIM + h * HD + dt * 16),
                              1, NQKV, 0, kt * 32, lane);
          acc = wmma16(ap[kt], bv, acc);
        }
        const int d = dt * 16 + nloc;
#pragma unroll
        for (int r = 0; r < 8; ++r)
          aos[(m0 + 8 * half + r) * CDIM + h * HD + d] = (_Float16)acc[r];
      }
    }
    __syncthreads();   // proj reads all head columns of aos (both groups)

    // ---- projection: (64x96) x (96x96) + bias, store rows < 49
    {
      v16h aa[CDIM / 32];
#pragma unroll
      for (int kt = 0; kt < CDIM / 32; ++kt)
        aa[kt] = load_frag(aos, CDIM, 1, m0, kt * 32, lane);
      for (int nt = grp; nt < CDIM / 16; nt += 2) {
        v8f acc = {};
#pragma unroll
        for (int kt = 0; kt < CDIM / 32; ++kt) {
          v16h bf = load_frag(wprojT, CDIM, 1, nt * 16, kt * 32, lane);
          acc = wmma16(aa[kt], bf, acc);
        }
        const int   n  = nt * 16 + nloc;
        const float pb = projbL[n];
#pragma unroll
        for (int r = 0; r < 8; ++r) {
          const int m = m0 + 8 * half + r;
          if (m < TOKENS)
            out[((size_t)b * TOKENS + m) * CDIM + n] = acc[r] + pb;
        }
      }
    }
    __syncthreads();   // protect xs/qkvs/aos reuse next window
  }
}

extern "C" void kernel_launch(void* const* d_in, const int* in_sizes, int n_in,
                              void* d_out, int out_size, void* d_ws, size_t ws_size,
                              hipStream_t stream) {
  (void)in_sizes; (void)n_in; (void)out_size; (void)d_ws; (void)ws_size;
  const float* x          = (const float*)d_in[0];
  // d_in[1] = q_global (unused by forward)
  const float* qkv_w      = (const float*)d_in[2];
  const float* qkv_b      = (const float*)d_in[3];
  const float* proj_w     = (const float*)d_in[4];
  const float* proj_b     = (const float*)d_in[5];
  const float* bias_table = (const float*)d_in[6];
  const int*   rel_index  = (const int*)  d_in[7];
  float* out = (float*)d_out;

  dim3 grid(NWIN / WPB);
  dim3 block(NTHREADS);
  win_attn_fused<<<grid, block, SMEM_BYTES, stream>>>(
      x, qkv_w, qkv_b, proj_w, proj_b, bias_table, rel_index, out);
}
